// SimilarityAttention_78331613544928
// MI455X (gfx1250) — matrix operand order repaired and checked by reading, not compiled, not run on hardware
//
#include <hip/hip_runtime.h>

#define DEVI __device__ __forceinline__

typedef unsigned short u16;
typedef unsigned int   u32;
typedef __attribute__((ext_vector_type(16))) __bf16 v16bf;
typedef __attribute__((ext_vector_type(8)))  float  v8f;
typedef __attribute__((ext_vector_type(4)))  u32    u32x4;

constexpr int CB = 4, CS = 2048, CD = 512, CH = 8, CP = 64, CHD = 64;

union ABf { v16bf v; u32x4 q[2]; };

DEVI u16 f32_bf16(float f) {
  u32 u = __float_as_uint(f);
  u = (u + 0x7FFFu + ((u >> 16) & 1u)) >> 16;
  return (u16)u;
}

DEVI v8f wmma_bf16(v16bf a, v16bf b, v8f c) {
  return __builtin_amdgcn_wmma_f32_16x16x32_bf16(false, a, false, b, (short)0, c,
                                                 false, false);
}

// A-operand frag (16x32 bf16): lane half h holds K = {8h..8h+7, 16+8h..23+8h}
DEVI v16bf frag_a(const u16* __restrict__ base, int ld, int k0) {
  int l = threadIdx.x & 31;
  int hf = l >> 4, m = l & 15;
  const u16* p = base + (size_t)m * ld + k0 + 8 * hf;
  ABf f;
  f.q[0] = *(const u32x4*)(p);
  f.q[1] = *(const u32x4*)(p + 16);
  return f.v;
}

// B-operand frag (32x16 bf16) from K-contiguous ("transposed") storage bt[n][k]
DEVI v16bf frag_b(const u16* __restrict__ bt, int ld, int n0, int k0) {
  int l = threadIdx.x & 31;
  int hf = l >> 4, n = l & 15;
  const u16* p = bt + (size_t)(n0 + n) * ld + k0 + 8 * hf;
  ABf f;
  f.q[0] = *(const u32x4*)(p);
  f.q[1] = *(const u32x4*)(p + 16);
  return f.v;
}

DEVI float hmax16(float v) {
  v = fmaxf(v, __shfl_xor(v, 1, 16));
  v = fmaxf(v, __shfl_xor(v, 2, 16));
  v = fmaxf(v, __shfl_xor(v, 4, 16));
  v = fmaxf(v, __shfl_xor(v, 8, 16));
  return v;
}
DEVI float hsum16(float v) {
  v += __shfl_xor(v, 1, 16);
  v += __shfl_xor(v, 2, 16);
  v += __shfl_xor(v, 4, 16);
  v += __shfl_xor(v, 8, 16);
  return v;
}

// ---------------- fused converts / weight transposes ---------------------------
__global__ __launch_bounds__(256) void k_prep(
    const float* __restrict__ q, const float* __restrict__ k,
    const float* __restrict__ v, const float* __restrict__ qpe,
    const float* __restrict__ kpe, const float* __restrict__ Wq,
    const float* __restrict__ Wk, const float* __restrict__ Wv,
    const float* __restrict__ Wo, u16* __restrict__ qb, u16* __restrict__ kb,
    u16* __restrict__ vb, u16* __restrict__ qpeb, u16* __restrict__ kpeb,
    u16* __restrict__ WqT, u16* __restrict__ WkT, u16* __restrict__ WvT,
    u16* __restrict__ WoT) {
  const int nBSD = CB * CS * CD, nBSP = CB * CS * CP, nDD = CD * CD;
  const int total = 3 * nBSD + 2 * nBSP + 4 * nDD;
  for (int i = blockIdx.x * blockDim.x + threadIdx.x; i < total;
       i += gridDim.x * blockDim.x) {
    int x = i;
    if (x < nBSD) { qb[x] = f32_bf16(q[x]); continue; }
    x -= nBSD;
    if (x < nBSD) { kb[x] = f32_bf16(k[x]); continue; }
    x -= nBSD;
    if (x < nBSD) { vb[x] = f32_bf16(v[x]); continue; }
    x -= nBSD;
    if (x < nBSP) { qpeb[x] = f32_bf16(qpe[x]); continue; }
    x -= nBSP;
    if (x < nBSP) { kpeb[x] = f32_bf16(kpe[x]); continue; }
    x -= nBSP;
    int kk = (x % nDD) / CD, nn = (x % nDD) % CD;
    size_t to = (size_t)nn * CD + kk;
    if (x < nDD) { WqT[to] = f32_bf16(Wq[x]); continue; }
    x -= nDD;
    if (x < nDD) { WkT[to] = f32_bf16(Wk[x]); continue; }
    x -= nDD;
    if (x < nDD) { WvT[to] = f32_bf16(Wv[x]); continue; }
    x -= nDD;
    WoT[to] = f32_bf16(Wo[x]);
  }
}

// ---------------- GEMM core: 32x64 per wave, double-buffered -------------------
DEVI void gemm_core(const u16* __restrict__ A, const u16* __restrict__ Bt,
                    int m0, int n0, v8f (&acc)[2][4]) {
  const u16* Ar0 = A + (size_t)m0 * CD;
  const u16* Ar1 = A + (size_t)(m0 + 16) * CD;
  v16bf af0 = frag_a(Ar0, CD, 0), af1 = frag_a(Ar1, CD, 0);
  v16bf bf0 = frag_b(Bt, CD, n0, 0), bf1 = frag_b(Bt, CD, n0 + 16, 0);
  v16bf bf2 = frag_b(Bt, CD, n0 + 32, 0), bf3 = frag_b(Bt, CD, n0 + 48, 0);
  for (int k0 = 0; k0 < CD; k0 += 32) {
    int kn = (k0 + 32) & (CD - 1);
    v16bf an0 = frag_a(Ar0, CD, kn), an1 = frag_a(Ar1, CD, kn);
    v16bf bn0 = frag_b(Bt, CD, n0, kn), bn1 = frag_b(Bt, CD, n0 + 16, kn);
    v16bf bn2 = frag_b(Bt, CD, n0 + 32, kn), bn3 = frag_b(Bt, CD, n0 + 48, kn);
    acc[0][0] = wmma_bf16(af0, bf0, acc[0][0]);
    acc[0][1] = wmma_bf16(af0, bf1, acc[0][1]);
    acc[0][2] = wmma_bf16(af0, bf2, acc[0][2]);
    acc[0][3] = wmma_bf16(af0, bf3, acc[0][3]);
    acc[1][0] = wmma_bf16(af1, bf0, acc[1][0]);
    acc[1][1] = wmma_bf16(af1, bf1, acc[1][1]);
    acc[1][2] = wmma_bf16(af1, bf2, acc[1][2]);
    acc[1][3] = wmma_bf16(af1, bf3, acc[1][3]);
    af0 = an0; af1 = an1;
    bf0 = bn0; bf1 = bn1; bf2 = bn2; bf3 = bn3;
  }
}

// normal output: C[m][n] bf16, LDS-coalesced b128 stores
__global__ __launch_bounds__(256) void k_gemm(const u16* __restrict__ A,
                                              const u16* __restrict__ Bt,
                                              u16* __restrict__ C, float alpha) {
  __shared__ __align__(16) u16 cbuf[8][32][64];  // 32 KB
  int wave = threadIdx.x >> 5;
  int mi = wave >> 2, ni = wave & 3;
  int m0 = blockIdx.x * 64 + mi * 32;
  int n0 = blockIdx.y * 256 + ni * 64;
  v8f acc[2][4] = {};
  gemm_core(A, Bt, m0, n0, acc);
  int l = threadIdx.x & 31, hf = l >> 4, nc = l & 15;
#pragma unroll
  for (int mi2 = 0; mi2 < 2; mi2++)
#pragma unroll
    for (int j = 0; j < 4; j++)
#pragma unroll
      for (int r = 0; r < 8; r++)
        cbuf[wave][mi2 * 16 + r + 8 * hf][j * 16 + nc] =
            f32_bf16(acc[mi2][j][r] * alpha);
  asm volatile("s_wait_dscnt 0" ::: "memory");
  const u16* src = &cbuf[wave][l][0];
  u16* dst = C + (size_t)(m0 + l) * CD + n0;
#pragma unroll
  for (int i = 0; i < 8; i++)
    *(u32x4*)(dst + i * 8) = *(const u32x4*)(src + i * 8);
}

// V projection: writes head-transposed Vt[b][h][d][s] directly (packed b128)
__global__ __launch_bounds__(256) void k_gemm_vt(const u16* __restrict__ A,
                                                 const u16* __restrict__ Bt,
                                                 u16* __restrict__ Vt) {
  int wave = threadIdx.x >> 5;
  int mi = wave >> 2, ni = wave & 3;
  int m0 = blockIdx.x * 64 + mi * 32;
  int n0 = blockIdx.y * 256 + ni * 64;
  v8f acc[2][4] = {};
  gemm_core(A, Bt, m0, n0, acc);
  int l = threadIdx.x & 31, hf = l >> 4, nc = l & 15;
#pragma unroll
  for (int mi2 = 0; mi2 < 2; mi2++)
#pragma unroll
    for (int j = 0; j < 4; j++) {
      int n = n0 + 16 * j + nc;
      int h = n >> 6, d = n & 63;
      int grow = m0 + mi2 * 16 + 8 * hf;   // 8-aligned, rows grow..grow+7
      int bb = grow >> 11, sidx = grow & (CS - 1);
      union { u32x4 qv; u16 e[8]; } pk;
#pragma unroll
      for (int r = 0; r < 8; r++) pk.e[r] = f32_bf16(acc[mi2][j][r]);
      *(u32x4*)(Vt + (((size_t)(bb * CH + h) * CHD + d) * CS + sidx)) = pk.qv;
    }
}

// -------- positional similarity: recompute 2-pass, p' = exp(s-max), Z = sum ----
__global__ __launch_bounds__(256) void k_simi(const u16* __restrict__ qpe,
                                              const u16* __restrict__ kpe,
                                              float* __restrict__ Pp,
                                              float* __restrict__ Zb) {
  int b = blockIdx.y;
  int wave = threadIdx.x >> 5;
  int q0 = blockIdx.x * 128 + wave * 16;
  const u16* qb = qpe + (size_t)b * CS * CP;
  const u16* kb = kpe + (size_t)b * CS * CP;
  float* prow = Pp + ((size_t)b * CS + q0) * CS;
  int l = threadIdx.x & 31, hf = l >> 4, nc = l & 15;

  v16bf a0 = frag_a(qb + (size_t)q0 * CP, CP, 0);
  v16bf a1 = frag_a(qb + (size_t)q0 * CP, CP, 32);

  float mx[8];
#pragma unroll
  for (int r = 0; r < 8; r++) mx[r] = -1e30f;

  // pass 1: row max only (no global round-trip; kpe is L2-resident)
  v16bf b0 = frag_b(kb, CP, 0, 0), b1 = frag_b(kb, CP, 0, 32);
  for (int t = 0; t < CS / 16; t++) {
    int cn = (t + 1 < CS / 16) ? (t + 1) * 16 : 0;
    v16bf nb0 = frag_b(kb, CP, cn, 0), nb1 = frag_b(kb, CP, cn, 32);
    v8f s = {};
    s = wmma_bf16(a0, b0, s);
    s = wmma_bf16(a1, b1, s);
#pragma unroll
    for (int r = 0; r < 8; r++) mx[r] = fmaxf(mx[r], s[r]);
    b0 = nb0; b1 = nb1;
  }
#pragma unroll
  for (int r = 0; r < 8; r++) mx[r] = hmax16(mx[r]);

  // pass 2: recompute, exponentiate, store p' once, accumulate Z
  float z[8];
#pragma unroll
  for (int r = 0; r < 8; r++) z[r] = 0.f;
  b0 = frag_b(kb, CP, 0, 0);
  b1 = frag_b(kb, CP, 0, 32);
  for (int t = 0; t < CS / 16; t++) {
    int c0 = t * 16;
    int cn = (t + 1 < CS / 16) ? c0 + 16 : 0;
    v16bf nb0 = frag_b(kb, CP, cn, 0), nb1 = frag_b(kb, CP, cn, 32);
    v8f s = {};
    s = wmma_bf16(a0, b0, s);
    s = wmma_bf16(a1, b1, s);
#pragma unroll
    for (int r = 0; r < 8; r++) {
      float p = __expf(s[r] - mx[r]);
      prow[(size_t)(r + 8 * hf) * CS + c0 + nc] = p;
      z[r] += p;
    }
    b0 = nb0; b1 = nb1;
  }
#pragma unroll
  for (int r = 0; r < 8; r++) z[r] = hsum16(z[r]);
  if (nc == 0) {
#pragma unroll
    for (int r = 0; r < 8; r++) Zb[(size_t)b * CS + q0 + r + 8 * hf] = z[r];
  }
}

// ---------------- flash attention with similarity modulation -------------------
// grid = (S/16, B); wave = head  => all 8 waves share the same Pp/mask/Z lines
__global__ __launch_bounds__(256) void k_attn(const u16* __restrict__ Qp,
                                              const u16* __restrict__ Kp,
                                              const u16* __restrict__ Vt,
                                              const float* __restrict__ Pp,
                                              const float* __restrict__ Zb,
                                              const int* __restrict__ mask,
                                              u16* __restrict__ AO) {
  __shared__ __align__(16) u16 pbuf[8][16][32];  // P-tile transpose staging, 8 KB
  __shared__ __align__(16) u16 obuf[8][16][64];  // output coalescing, 16 KB
  __shared__ __align__(16) int mLds[CS];         // mask row, 8 KB (async-loaded)
  int wave = threadIdx.x >> 5;  // == head
  int q0 = blockIdx.x * 16;
  int b = blockIdx.y;
  int h = wave;
  int l = threadIdx.x & 31, hf = l >> 4, nc = l & 15;

  const int* mb = mask + (size_t)b * CS;
  // stage the workgroup-shared mask row into LDS via the async copy path
  for (int i = threadIdx.x; i < CS / 4; i += 256) {
    u32 ldsoff = (u32)(uintptr_t)&mLds[i * 4];
    const int* ga = mb + i * 4;
    asm volatile("global_load_async_to_lds_b128 %0, %1, off"
                 :: "v"(ldsoff), "v"(ga) : "memory");
  }

  const u16* Qb = Qp + ((size_t)b * CS + q0) * CD + h * CHD;
  const u16* Kb = Kp + (size_t)b * CS * CD + h * CHD;
  const u16* Vb = Vt + ((size_t)(b * CH + h)) * CHD * CS;
  const float* Pr = Pp + ((size_t)b * CS + q0) * CS;

  v16bf a0 = frag_a(Qb, CD, 0);  // Q pre-scaled by 1/sqrt(HD)
  v16bf a1 = frag_a(Qb, CD, 32);

  float invZ[8], am[8], al[8];
#pragma unroll
  for (int r = 0; r < 8; r++) {
    invZ[r] = 1.0f / Zb[(size_t)b * CS + q0 + r + 8 * hf];
    am[r] = -1e30f;
    al[r] = 0.f;
  }
  v8f acc[4] = {};

  asm volatile("s_wait_asynccnt 0" ::: "memory");
  __syncthreads();

  for (int blk = 0; blk < CS / 32; blk++) {
    int c0 = blk * 32;
    if (blk + 1 < CS / 32) {
      // one prefetch covers the whole next simi tile (16 rows x 2 halves)
      __builtin_prefetch(Pr + (size_t)(l & 15) * CS + c0 + 32 + 16 * hf, 0, 3);
      __builtin_prefetch(Kb + (size_t)(c0 + 32 + nc) * CD, 0, 3);
    }
    // all operand loads issued up-front; softmax VALU chain overlaps them
    v16bf k0a = frag_b(Kb, CD, c0, 0), k0b = frag_b(Kb, CD, c0, 32);
    v16bf k1a = frag_b(Kb, CD, c0 + 16, 0), k1b = frag_b(Kb, CD, c0 + 16, 32);
    v16bf vf0 = frag_b(Vb, CS, 0, c0), vf1 = frag_b(Vb, CS, 16, c0);
    v16bf vf2 = frag_b(Vb, CS, 32, c0), vf3 = frag_b(Vb, CS, 48, c0);
    float sim0[8], sim1[8];
#pragma unroll
    for (int r = 0; r < 8; r++) {
      sim0[r] = Pr[(size_t)(r + 8 * hf) * CS + c0 + nc] * invZ[r];
      sim1[r] = Pr[(size_t)(r + 8 * hf) * CS + c0 + 16 + nc] * invZ[r];
    }
    float mt0 = (float)(mLds[c0 + nc] - 1) * 1e8f;
    float mt1 = (float)(mLds[c0 + 16 + nc] - 1) * 1e8f;

    v8f s0 = {}, s1 = {};
    s0 = wmma_bf16(a0, k0a, s0);
    s0 = wmma_bf16(a1, k0b, s0);
    s1 = wmma_bf16(a0, k1a, s1);
    s1 = wmma_bf16(a1, k1b, s1);

    float corr[8], p0[8], p1[8];
#pragma unroll
    for (int r = 0; r < 8; r++) {
      float x0 = s0[r] * sim0[r] + mt0;
      float x1 = s1[r] * sim1[r] + mt1;
      float tm = hmax16(fmaxf(x0, x1));
      float mn = fmaxf(am[r], tm);
      corr[r] = __expf(am[r] - mn);
      am[r] = mn;
      p0[r] = __expf(x0 - mn);
      p1[r] = __expf(x1 - mn);
      al[r] = al[r] * corr[r] + hsum16(p0[r] + p1[r]);
    }
#pragma unroll
    for (int j = 0; j < 4; j++)
#pragma unroll
      for (int r = 0; r < 8; r++) acc[j][r] *= corr[r];

#pragma unroll
    for (int r = 0; r < 8; r++) {
      pbuf[wave][r + 8 * hf][nc] = f32_bf16(p0[r]);
      pbuf[wave][r + 8 * hf][16 + nc] = f32_bf16(p1[r]);
    }
    asm volatile("s_wait_dscnt 0" ::: "memory");
    ABf fa;
    const u16* pp = &pbuf[wave][l & 15][hf * 8];
    fa.q[0] = *(const u32x4*)(pp);
    fa.q[1] = *(const u32x4*)(pp + 16);
    acc[0] = wmma_bf16(fa.v, vf0, acc[0]);
    acc[1] = wmma_bf16(fa.v, vf1, acc[1]);
    acc[2] = wmma_bf16(fa.v, vf2, acc[2]);
    acc[3] = wmma_bf16(fa.v, vf3, acc[3]);
  }

#pragma unroll
  for (int r = 0; r < 8; r++) {
    float il = 1.0f / al[r];
#pragma unroll
    for (int j = 0; j < 4; j++)
      obuf[wave][r + 8 * hf][j * 16 + nc] = f32_bf16(acc[j][r] * il);
  }
  asm volatile("s_wait_dscnt 0" ::: "memory");
  int row = l >> 1, half = l & 1;
  const u16* srcp = &obuf[wave][row][half * 32];
  u16* dst = AO + ((size_t)b * CS + q0 + row) * CD + h * CHD + half * 32;
#pragma unroll
  for (int i = 0; i < 4; i++)
    *(u32x4*)(dst + i * 8) = *(const u32x4*)(srcp + i * 8);
}

// ---------------- Wo GEMM + residual + LayerNorm -------------------------------
__global__ __launch_bounds__(256) void k_final(const u16* __restrict__ AO,
                                               const u16* __restrict__ WoT,
                                               const float* __restrict__ resid,
                                               const float* __restrict__ gamma,
                                               const float* __restrict__ beta,
                                               float* __restrict__ out) {
  __shared__ float xb[16][CD];  // 32 KB
  int q0 = blockIdx.x * 16;
  int wave = threadIdx.x >> 5;
  int n0 = wave * 64;
  const u16* Ar = AO + (size_t)q0 * CD;
  v8f acc[4] = {};
  v16bf af = frag_a(Ar, CD, 0);
  v16bf bf0 = frag_b(WoT, CD, n0, 0), bf1 = frag_b(WoT, CD, n0 + 16, 0);
  v16bf bf2 = frag_b(WoT, CD, n0 + 32, 0), bf3 = frag_b(WoT, CD, n0 + 48, 0);
  for (int k0 = 0; k0 < CD; k0 += 32) {
    int kn = (k0 + 32) & (CD - 1);
    v16bf an = frag_a(Ar, CD, kn);
    v16bf bn0 = frag_b(WoT, CD, n0, kn), bn1 = frag_b(WoT, CD, n0 + 16, kn);
    v16bf bn2 = frag_b(WoT, CD, n0 + 32, kn), bn3 = frag_b(WoT, CD, n0 + 48, kn);
    acc[0] = wmma_bf16(af, bf0, acc[0]);
    acc[1] = wmma_bf16(af, bf1, acc[1]);
    acc[2] = wmma_bf16(af, bf2, acc[2]);
    acc[3] = wmma_bf16(af, bf3, acc[3]);
    af = an; bf0 = bn0; bf1 = bn1; bf2 = bn2; bf3 = bn3;
  }
  int l = threadIdx.x & 31, hf = l >> 4, nc = l & 15;
#pragma unroll
  for (int j = 0; j < 4; j++)
#pragma unroll
    for (int r = 0; r < 8; r++) {
      int m = r + 8 * hf, n = n0 + 16 * j + nc;
      xb[m][n] = acc[j][r] + resid[(size_t)(q0 + m) * CD + n];
    }
  __syncthreads();

  int row = threadIdx.x >> 4, c = threadIdx.x & 15;
  float sum = 0.f, sq = 0.f;
#pragma unroll 4
  for (int i = 0; i < 32; i++) {
    float v = xb[row][c * 32 + i];
    sum += v;
    sq += v * v;
  }
  sum = hsum16(sum);
  sq = hsum16(sq);
  float mean = sum * (1.0f / CD);
  float var = sq * (1.0f / CD) - mean * mean;
  float rstd = rsqrtf(var + 1e-5f);
  float* orow = out + (size_t)(q0 + row) * CD;
#pragma unroll 4
  for (int i = 0; i < 32; i++) {
    int n = c * 32 + i;
    orow[n] = (xb[row][n] - mean) * rstd * gamma[n] + beta[n];
  }
}

// ---------------- host side ----------------------------------------------------
extern "C" void kernel_launch(void* const* d_in, const int* in_sizes, int n_in,
                              void* d_out, int out_size, void* d_ws,
                              size_t ws_size, hipStream_t stream) {
  const float* q_f = (const float*)d_in[0];
  const float* k_f = (const float*)d_in[1];
  const float* v_f = (const float*)d_in[2];
  const float* qpe_f = (const float*)d_in[3];
  const float* kpe_f = (const float*)d_in[4];
  const int* mask = (const int*)d_in[5];
  const float* Wq = (const float*)d_in[6];
  const float* Wk = (const float*)d_in[7];
  const float* Wv = (const float*)d_in[8];
  const float* Wo = (const float*)d_in[9];
  const float* gamma = (const float*)d_in[10];
  const float* beta = (const float*)d_in[11];
  float* out = (float*)d_out;

  const size_t nBSD = (size_t)CB * CS * CD;
  const size_t nBSP = (size_t)CB * CS * CP;
  const size_t nDD = (size_t)CD * CD;
  const size_t nBSS = (size_t)CB * CS * CS;
  const size_t nBS = (size_t)CB * CS;

  char* w = (char*)d_ws;
  auto alloc = [&](size_t bytes) {
    char* p = w;
    w += (bytes + 255) & ~(size_t)255;
    return p;
  };
  u16* qb = (u16*)alloc(nBSD * 2);
  u16* kb = (u16*)alloc(nBSD * 2);
  u16* vb = (u16*)alloc(nBSD * 2);
  u16* qpeb = (u16*)alloc(nBSP * 2);
  u16* kpeb = (u16*)alloc(nBSP * 2);
  u16* WqT = (u16*)alloc(nDD * 2);
  u16* WkT = (u16*)alloc(nDD * 2);
  u16* WvT = (u16*)alloc(nDD * 2);
  u16* WoT = (u16*)alloc(nDD * 2);
  u16* Qp = (u16*)alloc(nBSD * 2);
  u16* Kp = (u16*)alloc(nBSD * 2);
  u16* Vt = (u16*)alloc(nBSD * 2);
  float* Pp = (float*)alloc(nBSS * 4);
  float* Zb = (float*)alloc(nBS * 4);
  u16* AO = (u16*)alloc(nBSD * 2);
  (void)ws_size; (void)in_sizes; (void)n_in; (void)out_size;

  // fused fp32->bf16 converts + weight transposes (1 launch)
  k_prep<<<4096, 256, 0, stream>>>(q_f, k_f, v_f, qpe_f, kpe_f, Wq, Wk, Wv, Wo,
                                   qb, kb, vb, qpeb, kpeb, WqT, WkT, WvT, WoT);

  // projections (Q pre-scaled by 1/sqrt(HD)); V written head-transposed
  dim3 ggrid(CB * CS / 64, CD / 256);
  k_gemm<<<ggrid, 256, 0, stream>>>(qb, WqT, Qp, 0.125f);
  k_gemm<<<ggrid, 256, 0, stream>>>(kb, WkT, Kp, 1.0f);
  k_gemm_vt<<<ggrid, 256, 0, stream>>>(vb, WvT, Vt);

  // positional similarity (recompute 2-pass: p' + row sums, single 64MB write)
  k_simi<<<dim3(CS / 128, CB), 256, 0, stream>>>(qpeb, kpeb, Pp, Zb);

  // flash attention: one workgroup = (b, q-strip) x all 8 heads (wave = head)
  k_attn<<<dim3(CS / 16, CB), 256, 0, stream>>>(Qp, Kp, Vt, Pp, Zb, mask, AO);

  // output projection + residual + LayerNorm
  k_final<<<CB * CS / 16, 256, 0, stream>>>(AO, WoT, q_f, gamma, beta, out);
}
